// PCCNN_82978768159049
// MI455X (gfx1250) — matrix operand order repaired and checked by reading, not compiled, not run on hardware
//
#include <hip/hip_runtime.h>
#include <math.h>

// ---------------------------------------------------------------------------
// MI455X (gfx1250, wave32) implementation of the PCCNN reference.
//
// Heavy motifs (weight-MLP final layer, pcconv einsum) run as f16-in/f32-acc
// GEMMs on v_wmma_f32_16x16x32_f16. The B operand is kept TRANSPOSED in
// global memory (Bt[n][k]) so both A and B tiles stage identically and both
// fragments load as contiguous ds_load_b128 pairs (no scalar LDS gathers).
// Tiles are staged global->LDS with the Tensor Data Mover when the builtin is
// available (tensor_load_to_lds + s_wait_tensorcnt), else with per-thread
// b128 loads. Fragment layouts used here (A/B k order contiguous, not the ISA table order):
//   A 16x32 f16 : lane L row M=L&15; half h -> K=(L>>4)*16+h  (contig in As)
//   B 32x16 f16 : lane L col N=L&15; half t -> K=(L>>4)*16+t  (contig in Bt)
//   C 16x16 f32 : lane L col N=L&15; vgpr r -> M=r (L<16) / 8+r (L>=16)
// ---------------------------------------------------------------------------

typedef __attribute__((ext_vector_type(16))) _Float16 v16h;
typedef __attribute__((ext_vector_type(8)))  float    v8f;
typedef unsigned int uint32;
typedef uint32 v4u  __attribute__((ext_vector_type(4)));
typedef int    v8i_t __attribute__((ext_vector_type(8)));
typedef int    v4i_t __attribute__((ext_vector_type(4)));

#define TILE_M 128
#define TILE_N 128
#define TILE_K 32
#define LDS_PITCH 40   // 32 data halves + 8 pad halves (4 dwords) per row

#if defined(__has_builtin)
#if __has_builtin(__builtin_amdgcn_tensor_load_to_lds) && \
    __has_builtin(__builtin_amdgcn_s_wait_tensorcnt)
#define PCCNN_USE_TDM 1
#endif
#endif
#ifndef PCCNN_USE_TDM
#define PCCNN_USE_TDM 0
#endif

#if PCCNN_USE_TDM
// Build D# groups (CDNA5 ISA) for a 2D tile of
// 16-bit data: tile_dim0=32 elems, tile_dim1=128 rows, row stride = ld elems,
// LDS padding 4 dwords every 16 dwords (matches LDS_PITCH).
__device__ __forceinline__ void tdm_issue_tile(uint32 lds_addr, const void* gaddr,
                                               uint32 ld_elems)
{
    unsigned long long ga = (unsigned long long)gaddr;
    uint32 d0[4], d1[8];
    d0[0] = 1u;                                   // count=1, user descriptor
    d0[1] = lds_addr;                             // lds_addr  [63:32]
    d0[2] = (uint32)(ga & 0xffffffffu);           // global_addr[95:64]
    d0[3] = (uint32)((ga >> 32) & 0x01ffffffu)    // global_addr[120:96]
          | (2u << 30);                           // type=2 ("image")
    d1[0] = (1u << 16)                            // data_size = 2 bytes
          | (1u << 20)                            // pad_enable
          | (3u << 22)                            // pad_interval: 16 dwords
          | (3u << 25);                           // pad_amount : 4 dwords
    d1[1] = (32u & 0xffffu) << 16;                // tensor_dim0[15:0] = 32
    d1[2] = 0u | ((128u & 0xffffu) << 16);        // dim0 hi | tensor_dim1 lo =128
    d1[3] = 0u | (32u << 16);                     // dim1 hi | tile_dim0 = 32
    d1[4] = 128u;                                 // tile_dim1 = 128, tile_dim2=0
    d1[5] = ld_elems;                             // tensor_dim0_stride[31:0]
    d1[6] = 0u;                                   // stride hi | dim1_stride lo
    d1[7] = 0u;
    v4u  G0 = {d0[0], d0[1], d0[2], d0[3]};
    v8i_t G1 = {(int)d1[0], (int)d1[1], (int)d1[2], (int)d1[3],
                (int)d1[4], (int)d1[5], (int)d1[6], (int)d1[7]};
    v4i_t Z4 = (v4i_t)0;
#if __clang_major__ >= 23
    v8i_t Z8 = (v8i_t)0;
    __builtin_amdgcn_tensor_load_to_lds(G0, G1, Z4, Z4, Z8, 0);
#else
    __builtin_amdgcn_tensor_load_to_lds(G0, G1, Z4, Z4, 0);
#endif
}
#endif

// ---------------------------------------------------------------------------
// Unified WMMA GEMM: C[128*gridY x N] = A[M x K] * Bt[N x K]^T, f32 accum.
// emode 0: feature epilogue  -> out32[row*ldOut + (n/Cout)*strideO + colBase + n%Cout]
// emode 1: weight epilogue   -> outB[(o*Cout + n%Cout... ) ] transposed scatter:
//          outB[(o*CoutW + k)*ldBigBt + (i*CinW + c)]  (f16, bias fused)
// ---------------------------------------------------------------------------
__global__ __launch_bounds__(256) void gemm_wmma_kern(
    const _Float16* __restrict__ A,  int lda,
    const _Float16* __restrict__ Bt, int ldbt,
    int ksteps, int emode,
    // emode 0 params
    float* out32, int ldOut, int strideO, int colBase, int CoutF, int NvalidF,
    int accflag, int finalize, int doRelu, float scale,
    // emode 1 params
    const float* bias, int qiW, int CinW, int CoutW, int NvalidW, int pairsW,
    int ldBigBt, _Float16* outB)
{
    __shared__ __align__(16) _Float16 As[TILE_M][LDS_PITCH];
    __shared__ __align__(16) _Float16 Bs[TILE_N][LDS_PITCH];

    const int tid  = threadIdx.x;
    const int lane = tid & 31;
    const int wid  = tid >> 5;
    const int m0   = blockIdx.y * TILE_M;
    const int n0   = blockIdx.x * TILE_N;
    const int nw   = wid * 16;          // wave's 16-col slice of the N tile

    v8f acc[8];
#pragma unroll
    for (int m = 0; m < 8; ++m)
#pragma unroll
        for (int r = 0; r < 8; ++r) acc[m][r] = 0.0f;

#if PCCNN_USE_TDM
    const uint32 ldsA = (uint32)(size_t)&As[0][0];
    const uint32 ldsB = (uint32)(size_t)&Bs[0][0];
#endif

    for (int ks = 0; ks < ksteps; ++ks) {
        const int k0 = ks * TILE_K;

#if PCCNN_USE_TDM
        // Tensor Data Mover: DMA both 128x32-half tiles global->LDS (with the
        // +4-dword row padding applied by the TDM), one issuing wave.
        if (tid == 0) {
            tdm_issue_tile(ldsA, A  + (size_t)m0 * lda  + k0, (uint32)lda);
            tdm_issue_tile(ldsB, Bt + (size_t)n0 * ldbt + k0, (uint32)ldbt);
            __builtin_amdgcn_s_wait_tensorcnt(0);
        }
        __syncthreads();
#else
        // Fallback: per-thread b128 staging. 128 rows x 32 halves = 512 uint4
        // per tile, 2 per thread each.
#pragma unroll
        for (int it = 0; it < 2; ++it) {
            int idx = tid + it * 256;
            int row = idx >> 2;
            int c16 = (idx & 3) * 8;
            uint4 va = *(const uint4*)(A  + (size_t)(m0 + row) * lda  + k0 + c16);
            *(uint4*)&As[row][c16] = va;
            uint4 vb = *(const uint4*)(Bt + (size_t)(n0 + row) * ldbt + k0 + c16);
            *(uint4*)&Bs[row][c16] = vb;
        }
        __syncthreads();
        if (ks + 1 < ksteps) {
            __builtin_prefetch(A  + (size_t)(m0 + (tid >> 1)) * lda  + k0 + TILE_K, 0, 1);
            __builtin_prefetch(Bt + (size_t)(n0 + (tid >> 1)) * ldbt + k0 + TILE_K, 0, 1);
        }
#endif

        // --- B fragment: contiguous 16 halves (2x ds_load_b128) ---
        v16h bfrag;
        {
            const uint4* bp = (const uint4*)&Bs[nw + (lane & 15)][(lane >> 4) * 16];
            ((uint4*)&bfrag)[0] = bp[0];
            ((uint4*)&bfrag)[1] = bp[1];
        }
        // --- 8 WMMAs: full 128-row M sweep for this wave's N slice ---
#pragma unroll
        for (int m = 0; m < 8; ++m) {
            v16h afrag;
            int row = m * 16 + (lane & 15);
#pragma unroll
            for (int h = 0; h < 16; ++h) {
                int kk = (lane >> 4) * 16 + h;
                afrag[h] = As[row][kk];
            }
            acc[m] = __builtin_amdgcn_wmma_f32_16x16x32_f16(
                false, afrag, false, bfrag, (short)0, acc[m], false, false);
        }
        __syncthreads();
    }

    // ------------------------------ epilogue ------------------------------
    const int colT  = lane & 15;
    const int rbase = (lane >> 4) * 8;           // C layout: M = r (+8 hi lanes)
    const int gcol  = n0 + nw + colT;

#pragma unroll
    for (int m = 0; m < 8; ++m) {
#pragma unroll
        for (int r = 0; r < 8; ++r) {
            int grow = m0 + m * 16 + rbase + r;
            float v  = acc[m][r];
            if (emode == 0) {
                if (gcol < NvalidF) {
                    int o = gcol / CoutF, k = gcol % CoutF;
                    size_t off = (size_t)grow * ldOut + (size_t)o * strideO + colBase + k;
                    if (accflag)  v += out32[off];
                    if (finalize) { v *= scale; if (doRelu) v = v > 0.0f ? v : 0.0f; }
                    out32[off] = v;
                }
            } else {
                if (gcol < NvalidW && grow < pairsW) {
                    v += bias[gcol];
                    int c = gcol / CoutW, k = gcol % CoutW;
                    int o = grow / qiW,  i = grow % qiW;
                    // transposed scatter: row = output col (o,k), col = (i,c)
                    outB[(size_t)(o * CoutW + k) * ldBigBt + i * CinW + c] = (_Float16)v;
                }
            }
        }
    }
}

// ---------------------------------------------------------------------------
// Fourier features + the two small hidden layers of a kernel MLP.
// One block per (o,i) pair (padded to 1024 rows); writes f16 A operand of the
// weight GEMM, zero-padded in K.
// ---------------------------------------------------------------------------
__global__ __launch_bounds__(256) void wmlp_hidden(
    const float* __restrict__ angO, const float* __restrict__ angI,
    int qo, int qi, int lnum, int has_tap, float tap,
    const float* __restrict__ W0, const float* __restrict__ b0, int din, int h0d,
    const float* __restrict__ W1, const float* __restrict__ b1, int h1d,
    _Float16* __restrict__ H, int ldH)
{
    __shared__ float enc[96];
    __shared__ float h0s[256];
    __shared__ float h1s[256];

    const int row = blockIdx.x;
    const int t   = threadIdx.x;

    if (row >= qo * qi) {                      // M padding rows
        for (int j = t; j < ldH; j += 256)
            H[(size_t)row * ldH + j] = (_Float16)0.0f;
        return;
    }
    const int o = row / qi, i = row % qi;
    const float dx = angO[o*3+0] - angI[i*3+0];
    const float dy = angO[o*3+1] - angI[i*3+1];
    const float dz = angO[o*3+2] - angI[i*3+2];

    const int twoL = 2 * lnum;
    for (int e = t; e < 6 * lnum; e += 256) {
        int a = e / twoL, rr = e % twoL;
        int l = rr % lnum;
        float d   = (a == 0) ? dx : ((a == 1) ? dy : dz);
        float ang = d * 3.14159265358979323846f * exp2f((float)l);
        enc[e] = (rr < lnum) ? __sinf(ang) : __cosf(ang);
    }
    if (has_tap && t == 0) enc[6 * lnum] = tap;
    __syncthreads();

    const int indim = 6 * lnum + (has_tap ? 1 : 0);
    for (int j = t; j < h0d; j += 256) {
        float s = b0[j];
        for (int d = 0; d < indim; ++d) s += enc[d] * W0[d * h0d + j];
        h0s[j] = s > 0.0f ? s : 0.0f;
    }
    __syncthreads();
    for (int j = t; j < h1d; j += 256) {
        float s = b1[j];
        for (int d = 0; d < h0d; ++d) s += h0s[d] * W1[d * h1d + j];
        h1s[j] = s > 0.0f ? s : 0.0f;
    }
    __syncthreads();
    for (int j = t; j < ldH; j += 256)
        H[(size_t)row * ldH + j] = (j < h1d) ? (_Float16)h1s[j] : (_Float16)0.0f;
    (void)din;
}

// final-layer weights f32[h1 x dout] -> TRANSPOSED f16[npad x kpad], zero pad
__global__ void cvt_w(const float* __restrict__ W, int h1, int dout,
                      _Float16* __restrict__ O, int kpad, int npad)
{
    int idx = blockIdx.x * 256 + threadIdx.x;
    if (idx >= kpad * npad) return;
    int n = idx / kpad, k = idx % kpad;
    O[idx] = (k < h1 && n < dout) ? (_Float16)W[(size_t)k * dout + n] : (_Float16)0.0f;
}

// f32 feature buffer -> f16 A operand (128 padded rows, Kpad cols),
// optional spatial shift (factorised conv taps), zero padding.
__global__ void build_a(const float* __restrict__ F, int ldF, int kvalid,
                        _Float16* __restrict__ A, int lda, int axis, int off)
{
    int col = blockIdx.x * 256 + threadIdx.x;
    int row = blockIdx.y;
    if (col >= lda) return;
    float v = 0.0f;
    if (row < 125 && col < kvalid) {
        int src = row;
        if (axis >= 0) {
            int x = row / 25, y = (row / 5) % 5, z = row % 5;
            int c[3] = {x, y, z};
            c[axis] += off;
            if (c[axis] >= 0 && c[axis] < 5) src = c[0] * 25 + c[1] * 5 + c[2];
            else src = -1;
        }
        if (src >= 0) v = F[(size_t)src * ldF + col];
    }
    A[(size_t)row * lda + col] = (_Float16)v;
}

// f = relu(f_new + f_resid)
__global__ void resid_relu(const float* __restrict__ a, const float* __restrict__ b,
                           float* __restrict__ o, int n)
{
    int i = blockIdx.x * 256 + threadIdx.x;
    if (i >= n) return;
    float v = a[i] + b[i];
    o[i] = v > 0.0f ? v : 0.0f;
}

__global__ void copy_out(const float* __restrict__ F, int ldF, float* __restrict__ out)
{
    int i = blockIdx.x * 256 + threadIdx.x;
    if (i >= 125 * 30) return;
    out[i] = F[(size_t)(i / 30) * ldF + (i % 30)];
}

// ---------------------------------------------------------------------------
// Host orchestration
// ---------------------------------------------------------------------------
struct Mlp3 { const float *W0,*b0,*W1,*b1,*W2,*b2; int din,h0,h1,dout; };

static Mlp3 takeMlp(void* const* d_in, int& idx, int din, int h0, int h1, int dout)
{
    Mlp3 m;
    m.W0 = (const float*)d_in[idx++]; m.b0 = (const float*)d_in[idx++];
    m.W1 = (const float*)d_in[idx++]; m.b1 = (const float*)d_in[idx++];
    m.W2 = (const float*)d_in[idx++]; m.b2 = (const float*)d_in[idx++];
    m.din = din; m.h0 = h0; m.h1 = h1; m.dout = dout;
    return m;
}

static inline int pad128(int x) { return (x + 127) & ~127; }
static inline int pad32(int x)  { return (x + 31)  & ~31;  }

extern "C" void kernel_launch(void* const* d_in, const int* in_sizes, int n_in,
                              void* d_out, int out_size, void* d_ws, size_t ws_size,
                              hipStream_t stream)
{
    (void)in_sizes; (void)n_in; (void)out_size;
    const float* dmri = (const float*)d_in[0];
    const float* bin  = (const float*)d_in[1];
    const float* bout = (const float*)d_in[2];

    // Param leaves assumed in jax pytree (sorted-key dict) order:
    // in_blocks, out_blocks, pc_in, pc_mid, pc_out1, pc_out2
    int idx = 3;
    Mlp3 ib_c1[3], ib_c3[3][3], ob_c1[5], ob_c3[5][3];
    for (int n = 0; n < 3; ++n) {
        ib_c1[n]    = takeMlp(d_in, idx, 30, 128, 32, 256 * 128);
        ib_c3[n][0] = takeMlp(d_in, idx, 31, 128, 32, 256 * 32);
        ib_c3[n][1] = takeMlp(d_in, idx, 31, 128, 32, 32 * 32);
        ib_c3[n][2] = takeMlp(d_in, idx, 31, 128, 32, 32 * 32);
    }
    for (int n = 0; n < 5; ++n) {
        ob_c1[n]    = takeMlp(d_in, idx, 90, 16, 16, 160 * 128);
        ob_c3[n][0] = takeMlp(d_in, idx, 91, 16, 16, 160 * 32);
        ob_c3[n][1] = takeMlp(d_in, idx, 91, 16, 16, 32 * 32);
        ob_c3[n][2] = takeMlp(d_in, idx, 91, 16, 16, 32 * 32);
    }
    Mlp3 pc_in   = takeMlp(d_in, idx, 90, 64, 64, 256);
    Mlp3 pc_mid  = takeMlp(d_in, idx, 90, 32, 256, 256 * 160);
    Mlp3 pc_out1 = takeMlp(d_in, idx, 90, 32, 64, 160 * 64);
    Mlp3 pc_out2 = takeMlp(d_in, idx, 60, 256, 256, 64);

    // ---- workspace carve ----
    char* w = (char*)d_ws;
    auto alloc = [&](size_t bytes) -> void* {
        void* p = (void*)w; w += (bytes + 255) & ~(size_t)255; return p;
    };
    _Float16* A16  = (_Float16*)alloc((size_t)128 * 7680 * 2);   // feature A operand
    _Float16* H16  = (_Float16*)alloc((size_t)1024 * 256 * 2);   // MLP hidden (A of wgemm)
    _Float16* W16t = (_Float16*)alloc((size_t)40960 * 256 * 2);  // final weights, [n][k]
    _Float16* B16t = (_Float16*)alloc((size_t)4864 * 7680 * 2);  // generated weights, [n][k]
    float* cur = (float*)alloc((size_t)128 * 7680 * 4);
    float* tmp = (float*)alloc((size_t)128 * 7680 * 4);
    float* nxt = (float*)alloc((size_t)128 * 7680 * 4);
    float* g0  = (float*)alloc((size_t)128 * 960 * 4);
    float* g1  = (float*)alloc((size_t)128 * 960 * 4);
    if ((size_t)(w - (char*)d_ws) > ws_size) return;   // not enough scratch

    // ---- generate the big weight operand for one conv (f16, L2-resident) ----
    // kfeatPad = padded K of the downstream feature GEMM = row pitch of B16t.
    auto wgen = [&](const Mlp3& m, const float* angO, const float* angI, int lnum,
                    int has_tap, float tap, int Cin, int Cout, int kfeatPad) {
        int kpadH   = pad32(m.h1);
        int noutPad = pad128(m.dout);
        wmlp_hidden<<<1024, 256, 0, stream>>>(angO, angI, 30, 30, lnum, has_tap, tap,
            m.W0, m.b0, m.din, m.h0, m.W1, m.b1, m.h1, H16, kpadH);
        int tot = kpadH * noutPad;
        cvt_w<<<(tot + 255) / 256, 256, 0, stream>>>(m.W2, m.h1, m.dout, W16t,
                                                     kpadH, noutPad);
        gemm_wmma_kern<<<dim3(noutPad / 128, 1024 / 128), 256, 0, stream>>>(
            H16, kpadH, W16t, kpadH, kpadH / 32, /*emode=*/1,
            nullptr, 0, 0, 0, 1, 0, 0, 0, 0, 0.0f,
            m.b2, 30, Cin, Cout, m.dout, 900, kfeatPad, B16t);
    };

    auto builda = [&](const float* F, int ldF, int Kvalid, int axis, int off) {
        int kpad = pad32(Kvalid);
        build_a<<<dim3((kpad + 255) / 256, 128), 256, 0, stream>>>(
            F, ldF, Kvalid, A16, kpad, axis, off);
    };

    auto fgemm = [&](int Kfeat, int NbigPad, int Cout, float* out32, int ldOut,
                     int strideO, int colBase, int accf, int fin, int relu) {
        int kpad = pad32(Kfeat);
        gemm_wmma_kern<<<dim3(NbigPad / 128, 1), 256, 0, stream>>>(
            A16, kpad, B16t, kpad, kpad / 32, /*emode=*/0,
            out32, ldOut, strideO, colBase, Cout, 30 * Cout, accf, fin, relu,
            1.0f / 30.0f,
            nullptr, 0, 0, 1, 0, 0, 0, nullptr);
    };

    auto conv_point = [&](const Mlp3& m, const float* angI, const float* angO, int lnum,
                          int Cin, int Cout, const float* srcF, int ldF, int Kfeat,
                          float* dst, int ldOut, int strideO, int colBase, int relu) {
        int NbigPad = pad128(30 * Cout);
        wgen(m, angO, angI, lnum, 0, 0.0f, Cin, Cout, pad32(Kfeat));
        builda(srcF, ldF, Kfeat, -1, 0);
        fgemm(Kfeat, NbigPad, Cout, dst, ldOut, strideO, colBase, 0, 1, relu);
    };

    auto conv_axis = [&](const Mlp3& m, const float* angI, const float* angO, int lnum,
                         int Cin, int Cout, const float* srcF, int ldF, int Kfeat,
                         float* dst, int ldOut, int strideO, int colBase,
                         int axis, int relu) {
        int NbigPad = pad128(30 * Cout);
        for (int p = 0; p < 3; ++p) {            // fixed length-3 kernel taps
            wgen(m, angO, angI, lnum, 1, (float)(p - 1), Cin, Cout, pad32(Kfeat));
            builda(srcF, ldF, Kfeat, axis, p - 1);
            fgemm(Kfeat, NbigPad, Cout, dst, ldOut, strideO, colBase,
                  p > 0, p == 2, relu);
        }
    };

    auto block = [&](const Mlp3& c1, const Mlp3 c3[3], const float* angI,
                     const float* angO, int lnum, int Cin, int out1, int out3,
                     int residual) {
        const int Cc = out1 + out3, ldC = 30 * Cc;
        // f1 = relu(point conv) -> tmp[:, o*Cc + 0..out1)
        conv_point(c1, angI, angO, lnum, Cin, out1, cur, 30 * Cin, 30 * Cin,
                   tmp, ldC, Cc, 0, 1);
        // f3 = relu(3x1x1 -> 1x3x1 -> 1x1x3), relu only on last leg
        conv_axis(c3[0], angI, angO, lnum, Cin,  out3, cur, 30 * Cin,  30 * Cin,
                  g0, 30 * out3, out3, 0, 0, 0);
        conv_axis(c3[1], angO, angO, lnum, out3, out3, g0, 30 * out3, 30 * out3,
                  g1, 30 * out3, out3, 0, 1, 0);
        conv_axis(c3[2], angO, angO, lnum, out3, out3, g1, 30 * out3, 30 * out3,
                  tmp, ldC, Cc, out1, 2, 1);
        if (residual) {
            int n = 128 * ldC;
            resid_relu<<<(n + 255) / 256, 256, 0, stream>>>(tmp, cur, nxt, n);
            float* t = cur; cur = nxt; nxt = t;
        } else {
            float* t = cur; cur = tmp; tmp = t;
        }
    };

    // ------------------------------- pipeline -------------------------------
    // pc_in: f = relu(pcconv(dmri[...,None])), Cin=1 -> 256
    conv_point(pc_in, bin, bin, 15, 1, 256, dmri, 30, 30, cur, 7680, 256, 0, 1);

    for (int n = 0; n < 3; ++n)
        block(ib_c1[n], ib_c3[n], bin, bin, 5, 256, 128, 128, 1);

    conv_point(pc_mid, bin, bin, 15, 256, 160, cur, 7680, 7680, tmp, 4800, 160, 0, 1);
    { float* t = cur; cur = tmp; tmp = t; }

    for (int n = 0; n < 5; ++n)
        block(ob_c1[n], ob_c3[n], (n == 0) ? bin : bout, bout, 15, 160, 128, 32,
              n != 0);

    conv_point(pc_out1, bout, bout, 15, 160, 64, cur, 4800, 4800, tmp, 1920, 64, 0, 1);
    { float* t = cur; cur = tmp; tmp = t; }

    conv_point(pc_out2, bout, bout, 10, 64, 1, cur, 1920, 1920, tmp, 30, 1, 0, 0);

    copy_out<<<(125 * 30 + 255) / 256, 256, 0, stream>>>(tmp, 30, (float*)d_out);
}
